// Lookup_94489281383
// MI455X (gfx1250) — matrix operand order repaired and checked by reading, not compiled, not run on hardware
//
#include <hip/hip_runtime.h>
#include <hip/hip_bf16.h>
#include <stdint.h>

// ---------------------------------------------------------------------------
// Problem geometry (fixed by the reference): C=256, H=W=96, P = H*W = 9216.
// corr GEMM: (9216 x 9216) = A^T * B with K=256, scaled by 1/sqrt(256)=1/16.
// Pyramid: 96x96 -> 48 -> 24 -> 12 target dims. Lookup: 41 diamond offsets,
// 4 scales, bilinear, zero padding, output (41, 4, 96, 96).
// ---------------------------------------------------------------------------

#define CDIM   256
#define HW     96
#define P      (HW * HW)          // 9216
#define PT     (P / 16)           // 576 16-row tiles
#define KT     (CDIM / 32)        // 8 k-steps of 32
#define LOOKN  41
#define NTB    8                  // n-tiles per block (128 columns)

typedef __bf16 v16bf __attribute__((ext_vector_type(16)));
typedef float  v8f   __attribute__((ext_vector_type(8)));
typedef int    v4i   __attribute__((ext_vector_type(4)));

// Pointer types matching the async global->LDS builtin prototype
// (int4 in AS(1) -> int4 in AS(3)).
typedef __attribute__((address_space(1))) v4i* gv4i_p;
typedef __attribute__((address_space(3))) v4i* sv4i_p;

__device__ __constant__ float c_dy[LOOKN] = {
  -4,
  -3,-3,-3,
  -2,-2,-2,-2,-2,
  -1,-1,-1,-1,-1,-1,-1,
   0, 0, 0, 0, 0, 0, 0, 0, 0,
   1, 1, 1, 1, 1, 1, 1,
   2, 2, 2, 2, 2,
   3, 3, 3,
   4
};
__device__ __constant__ float c_dx[LOOKN] = {
   0,
  -1, 0, 1,
  -2,-1, 0, 1, 2,
  -3,-2,-1, 0, 1, 2, 3,
  -4,-3,-2,-1, 0, 1, 2, 3, 4,
  -3,-2,-1, 0, 1, 2, 3,
  -2,-1, 0, 1, 2,
  -1, 0, 1,
   0
};

__device__ __forceinline__ __bf16 f2bf(float f) {
  union { float f; uint32_t u; } v; v.f = f;
  uint32_t r = v.u + 0x7FFFu + ((v.u >> 16) & 1u);   // round-to-nearest-even
  uint16_t h = (uint16_t)(r >> 16);
  return __builtin_bit_cast(__bf16, h);
}

// ---------------------------------------------------------------------------
// Pack fp32 channel-major features into bf16 WMMA fragment order.
// One thread per (tile16, ktile32, lane) -> 16 bf16 (32 B) contiguous.
// A (16x32, MxK) lane layout (ISA 7.12.2): lane<16: row=lane, K in
// {0..7,16..23}; lane>=16: row=lane-16, K in {8..15,24..31}.
// B (32x16, KxN): lane<16: col=lane, K in {0..7,16..23}; lane>=16: col=lane-16, K in {8..15,24..31}.
// ---------------------------------------------------------------------------
__global__ void __launch_bounds__(256)
pack_tiles(const float* __restrict__ src, __bf16* __restrict__ dst, int isA) {
  int idx = blockIdx.x * blockDim.x + threadIdx.x;     // tile*256 + kt*32 + lane
  if (idx >= PT * KT * 32) return;
  int lane = idx & 31;
  int kt   = (idx >> 5) & (KT - 1);
  int tile = idx >> 8;
  int m  = lane & 15;
  int hi = lane >> 4;
  int pb = tile * 16 + m;
  int kb = kt * 32;
  __bf16* o = dst + (size_t)idx * 16;
#pragma unroll
  for (int e = 0; e < 16; ++e) {
    int k = isA ? ((e & 7) + ((e >> 3) << 4) + (hi << 3))
                : ((e & 7) + ((e >> 3) << 4) + (hi << 3));
    o[e] = f2bf(src[(size_t)(kb + k) * P + pb]);
  }
}

// ---------------------------------------------------------------------------
// Correlation GEMM, block tile 128(M) x 128(N), 8 waves of 16(M) x 128(N).
// B fragments (64 KB, identical for all 8 waves) are staged once per block
// into LDS -- via gfx1250 async global->LDS when available -- cutting L2
// fragment traffic ~5x. A fragments stream per-wave from L2 (packed feats
// total 9.4 MB and stay L2-resident). 64 WMMAs per wave, K=256.
// ---------------------------------------------------------------------------
__global__ void __launch_bounds__(256)
corr_gemm(const __bf16* __restrict__ Ap, const __bf16* __restrict__ Bp,
          float* __restrict__ corr) {
  __shared__ __bf16 sB[NTB * KT * 32 * 16];   // 32768 bf16 = 64 KB

  const int tid   = threadIdx.x;
  const int lane  = tid & 31;
  const int wave  = tid >> 5;
  const int mTile = blockIdx.x * 8 + wave;    // 0..575
  const int nT0   = blockIdx.y * NTB;         // 0..568 step 8

  // ---- cooperative stage of block's B fragments into LDS (64 KB) ----
  {
    const uint4* g = (const uint4*)(Bp + (size_t)nT0 * (KT * 32 * 16));
    uint4*       s = (uint4*)sB;              // 4096 uint4 = 16 per thread
#if __has_builtin(__builtin_amdgcn_global_load_async_to_lds_b128)
#pragma unroll
    for (int it = 0; it < 16; ++it) {
      int idx = tid + it * 256;
      __builtin_amdgcn_global_load_async_to_lds_b128(
          (gv4i_p)(g + idx), (sv4i_p)(s + idx), 0, 0);
    }
#if __has_builtin(__builtin_amdgcn_s_wait_asynccnt)
    __builtin_amdgcn_s_wait_asynccnt(0);
#else
    asm volatile("s_wait_asynccnt 0x0" ::: "memory");
#endif
#else
#pragma unroll
    for (int it = 0; it < 16; ++it) {
      int idx = tid + it * 256;
      s[idx] = g[idx];
    }
#endif
  }
  __syncthreads();

  // ---- main K loop: 8 k-steps x 8 n-tiles of v_wmma_f32_16x16x32_bf16 ----
  v8f acc[NTB];
#pragma unroll
  for (int n = 0; n < NTB; ++n) acc[n] = (v8f){};

  const __bf16* aB  = Ap + (((size_t)mTile * KT) * 32 + lane) * 16;
  const __bf16* sBl = sB + lane * 16;

#pragma unroll
  for (int kt = 0; kt < KT; ++kt) {
    v16bf a = *(const v16bf*)(aB + (size_t)kt * 512);
    if (kt + 1 < KT)
      __builtin_prefetch(aB + (size_t)(kt + 1) * 512, 0, 0);
#pragma unroll
    for (int n = 0; n < NTB; ++n) {
      v16bf b = *(const v16bf*)(sBl + ((size_t)n * KT + kt) * 512);
      acc[n] = __builtin_amdgcn_wmma_f32_16x16x32_bf16(
          false, a, false, b, (short)0, acc[n], false, false);
    }
  }

  // ---- scale + store. C/D layout: VGPR r, lane l -> row r+8*(l>=16), col l&15
  const float sc = 0.0625f;                 // 1/sqrt(256)
  const int n0 = lane & 15;
  const int mo = (lane >> 4) * 8;
#pragma unroll
  for (int r = 0; r < 8; ++r) {
    size_t row  = (size_t)mTile * 16 + mo + r;
    float* base = corr + row * P + (size_t)nT0 * 16 + n0;
#pragma unroll
    for (int n = 0; n < NTB; ++n)
      base[n * 16] = acc[n][r] * sc;
  }
}

// ---------------------------------------------------------------------------
// 2x2 average pool over the last two (target) dims. in: (P, hin, hin),
// out: (P, hin/2, hin/2). One thread per output element.
// ---------------------------------------------------------------------------
__global__ void __launch_bounds__(256)
pool2(const float* __restrict__ in, float* __restrict__ out, int hin, int n) {
  int idx = blockIdx.x * blockDim.x + threadIdx.x;
  if (idx >= n) return;
  int hout = hin >> 1;
  int x = idx % hout;
  int t = idx / hout;
  int y = t % hout;
  int pix = t / hout;
  const float* p = in + ((size_t)pix * hin + 2 * y) * hin + 2 * x;
  out[idx] = 0.25f * (p[0] + p[1] + p[hin] + p[hin + 1]);
}

// ---------------------------------------------------------------------------
// Multi-scale lookup with the reference's swapped-axis grid_sample semantics:
//   ic = (i + flow0)/s + dy ; jc = (j + flow1)/s + dx
//   px = ic*(hk-1)/hk (width axis) ; py = jc*(hk-1)/hk (height axis)
// Bilinear, zero padding. out[l][k][i][j].
// ---------------------------------------------------------------------------
__global__ void __launch_bounds__(256)
lookup_kernel(const float* __restrict__ vol0, const float* __restrict__ vol1,
              const float* __restrict__ vol2, const float* __restrict__ vol3,
              const float* __restrict__ flow, float* __restrict__ out) {
  int idx = blockIdx.x * blockDim.x + threadIdx.x;
  if (idx >= LOOKN * 4 * P) return;
  int j = idx % HW;
  int t = idx / HW;
  int i = t % HW;
  t /= HW;
  int k = t & 3;
  int l = t >> 2;

  const float* vol = (k == 0) ? vol0 : (k == 1) ? vol1 : (k == 2) ? vol2 : vol3;
  int   hk = HW >> k;
  float s  = (float)(1 << k);

  float f0 = flow[i * HW + j];
  float f1 = flow[P + i * HW + j];
  float ic = ((float)i + f0) / s + c_dy[l];
  float jc = ((float)j + f1) / s + c_dx[l];
  float ratio = (float)(hk - 1) / (float)hk;
  float px = ic * ratio;    // samples width axis
  float py = jc * ratio;    // samples height axis

  float x0 = floorf(px), y0 = floorf(py);
  float wx = px - x0, wy = py - y0;
  int xi = (int)x0, yi = (int)y0;

  const float* v = vol + (size_t)(i * HW + j) * hk * hk;
  auto g = [&](int yy, int xx) -> float {
    if (yy < 0 || yy >= hk || xx < 0 || xx >= hk) return 0.0f;
    return v[yy * hk + xx];
  };

  float val = g(yi,     xi    ) * (1.0f - wy) * (1.0f - wx)
            + g(yi,     xi + 1) * (1.0f - wy) * wx
            + g(yi + 1, xi    ) * wy * (1.0f - wx)
            + g(yi + 1, xi + 1) * wy * wx;
  out[idx] = val;
}

// ---------------------------------------------------------------------------
// Host-side orchestration (graph-capture safe: only kernel launches).
// ---------------------------------------------------------------------------
extern "C" void kernel_launch(void* const* d_in, const int* in_sizes, int n_in,
                              void* d_out, int out_size, void* d_ws, size_t ws_size,
                              hipStream_t stream) {
  (void)in_sizes; (void)n_in; (void)out_size; (void)ws_size;
  const float* feat1 = (const float*)d_in[0];   // (256, 96, 96) fp32
  const float* feat2 = (const float*)d_in[1];   // (256, 96, 96) fp32
  const float* flow  = (const float*)d_in[2];   // (2, 96, 96)   fp32
  float* out = (float*)d_out;                   // (41, 4, 96, 96) fp32

  // Workspace carve-up (all offsets 256B-aligned by construction).
  char* ws = (char*)d_ws;
  const size_t packBytes = (size_t)P * CDIM * sizeof(__bf16);   // 4,718,592
  __bf16* Ap  = (__bf16*)ws;
  __bf16* Bp  = (__bf16*)(ws + packBytes);
  float* vol0 = (float*)(ws + 2 * packBytes);                   // (P, 96, 96)
  float* vol1 = vol0 + (size_t)P * 96 * 96;                     // (P, 48, 48)
  float* vol2 = vol1 + (size_t)P * 48 * 48;                     // (P, 24, 24)
  float* vol3 = vol2 + (size_t)P * 24 * 24;                     // (P, 12, 12)

  // 1) fp32 -> bf16 fragment packing (feat1 as A, feat2 as B).
  {
    int threads = PT * KT * 32;                 // 147456
    pack_tiles<<<(threads + 255) / 256, 256, 0, stream>>>(feat1, Ap, 1);
    pack_tiles<<<(threads + 255) / 256, 256, 0, stream>>>(feat2, Bp, 0);
  }

  // 2) WMMA correlation GEMM -> vol0. Block tile 128x128.
  {
    dim3 grid(PT / 8, P / (NTB * 16));          // (72, 72)
    corr_gemm<<<grid, 256, 0, stream>>>(Ap, Bp, vol0);
  }

  // 3) Pyramid pooling.
  {
    int n1 = P * 48 * 48;
    int n2 = P * 24 * 24;
    int n3 = P * 12 * 12;
    pool2<<<(n1 + 255) / 256, 256, 0, stream>>>(vol0, vol1, 96, n1);
    pool2<<<(n2 + 255) / 256, 256, 0, stream>>>(vol1, vol2, 48, n2);
    pool2<<<(n3 + 255) / 256, 256, 0, stream>>>(vol2, vol3, 24, n3);
  }

  // 4) Bilinear lookup at 4 scales -> (41, 4, 96, 96).
  {
    int n = LOOKN * 4 * P;                      // 1,511,424
    lookup_kernel<<<(n + 255) / 256, 256, 0, stream>>>(vol0, vol1, vol2, vol3,
                                                       flow, out);
  }
}
